// PoseMapFromCordinatesLayer_45191645888552
// MI455X (gfx1250) — matrix operand order repaired and checked by reading, not compiled, not run on hardware
//
#include <hip/hip_runtime.h>

typedef __attribute__((ext_vector_type(16))) _Float16 v16h;
typedef __attribute__((ext_vector_type(8)))  float    v8f;

#define MAP_H  128
#define MAP_W  64
#define KPTS   18
#define AHALF  5
#define AWHALF 5
#define ROWLEN (MAP_W * KPTS)    // 1152 floats per h-row, j = w*18 + k
#define PER_B  (MAP_H * ROWLEN)  // 147456 floats per batch element
#define NTILES (ROWLEN / 16)     // 72 j-tiles of 16 columns

// out[b,h,w,k] = (|h-r[b,k]|<=5) * (|w-c[b,k]|<=5)
// Computed as D = A x B per (b, 16-row h tile) with one f16 WMMA per 16x16 tile:
//   A[m,K] = row indicator for h = h0+m, keypoint K  (K<18, zero-padded to 32)
//   B[K,j] = (K == k(j)) * column indicator for w(j)
__global__ __launch_bounds__(128)
void pose_map_wmma_kernel(const int* __restrict__ x, float* __restrict__ out) {
  __shared__ int rs[KPTS];
  __shared__ int cs[KPTS];

  const int b  = blockIdx.x >> 3;   // 256 batches
  const int ht = blockIdx.x & 7;    // 8 h-tiles of 16 rows
  const int h0 = ht * 16;
  const int tid = threadIdx.x;

  if (tid < KPTS) {
    rs[tid] = x[b * (KPTS * 2) + tid * 2 + 0];
    cs[tid] = x[b * (KPTS * 2) + tid * 2 + 1];
  }
  __syncthreads();

  const int lane = tid & 31;
  const int wave = tid >> 5;
  const int hi   = lane >> 4;      // 0: lanes 0-15, 1: lanes 16-31
  const int mn   = lane & 15;      // A row within tile / B,D column within tile

  // ---- Build A tile (16x32 f16). ISA layout for 16-bit A 16x32:
  //   VGPR v<4 : K = 2v,2v+1  (+8 for lanes 16-31)
  //   VGPR v>=4: K = 2v+8,... (+8 for lanes 16-31)
  const int h = h0 + mn;
  v16h a;
#pragma unroll
  for (int e = 0; e < 16; ++e) {
    const int v = e >> 1;
    const int K = ((v < 4) ? (2 * v) : (2 * v + 8)) + (e & 1) + (hi ? 8 : 0);
    _Float16 val = (_Float16)0.0f;
    if (K < KPTS) {                 // K>=18 pads with zero (folds for e>=10)
      int d = h - rs[K];
      d = d < 0 ? -d : d;
      val = (d <= AHALF) ? (_Float16)1.0f : (_Float16)0.0f;
    }
    a[e] = val;
  }

  // Per-lane output base: column (j0 + mn), row block (h0 + 8*hi)
  float* const outb = out + (size_t)b * PER_B
                          + (size_t)(h0 + hi * 8) * ROWLEN + mn;

  // ---- 18 j-tiles per wave: build one-hot B column, 1 WMMA, 8 coalesced stores
  for (int t = wave; t < NTILES; t += 4) {
    const int j = t * 16 + mn;
    const int w = j / KPTS;
    const int k = j - w * KPTS;
    int d = w - cs[k];
    d = d < 0 ? -d : d;
    const _Float16 cind = (d <= AWHALF) ? (_Float16)1.0f : (_Float16)0.0f;

    // Dense 32x16 f16 B layout (matches A): element e<8 -> K=e+8*hi,
    // e>=8 -> K=e+8+8*hi. Column j has a single nonzero at K = k(j).
    const int kk = k - (hi ? 8 : 0);
    const int sel = (kk >= 0 && kk < 8) ? kk : ((kk >= 16 && kk < 24) ? kk - 8 : -1);
    v16h bm;
#pragma unroll
    for (int e = 0; e < 16; ++e)
      bm[e] = (e == sel) ? cind : (_Float16)0.0f;

    v8f c = {};
    c = __builtin_amdgcn_wmma_f32_16x16x32_f16(
        /*neg_a=*/false, a, /*neg_b=*/false, bm,
        /*c_mod=*/(short)0, c, /*reuse_a=*/false, /*reuse_b=*/false);

    // D layout: VGPR v -> row h0 + v + 8*hi, lane group -> contiguous columns.
    float* p = outb + t * 16;
#pragma unroll
    for (int v = 0; v < 8; ++v)
      p[(size_t)v * ROWLEN] = c[v];
  }
}

extern "C" void kernel_launch(void* const* d_in, const int* in_sizes, int n_in,
                              void* d_out, int out_size, void* d_ws, size_t ws_size,
                              hipStream_t stream) {
  const int* x = (const int*)d_in[0];     // (256, 18, 2) int32
  float* out   = (float*)d_out;           // (256, 128, 64, 18) float32
  dim3 grid(256 * 8);                     // one block per (b, h-tile)
  dim3 block(128);                        // 4 wave32s
  hipLaunchKernelGGL(pose_map_wmma_kernel, grid, block, 0, stream, x, out);
}
